// SimpleGraphConv_88218628260601
// MI455X (gfx1250) — matrix operand order repaired and checked by reading, not compiled, not run on hardware
//
#include <hip/hip_runtime.h>
#include <hip/hip_bf16.h>

typedef __attribute__((ext_vector_type(16))) __bf16 bf16x16;
typedef __attribute__((ext_vector_type(8)))  __bf16 bf16x8;
typedef __attribute__((ext_vector_type(8)))  float  f32x8;
typedef __attribute__((ext_vector_type(4)))  float  f32x4;

#define NN 8192   // nodes
#define DD 256    // feature dim (in == out)

// ---------------------------------------------------------------------------
// Kernel 1: d[i] = rsqrt( sum_j adj[i][j] + 1 )   (rowsum of A_hat = adj + I)
// One wave32 per row; 8 rows per 256-thread block.
// ---------------------------------------------------------------------------
__global__ __launch_bounds__(256) void gcn_rowsum_kernel(
    const float* __restrict__ adj, float* __restrict__ dvec) {
  const int wave = threadIdx.x >> 5;
  const int lane = threadIdx.x & 31;
  const int row  = blockIdx.x * 8 + wave;
  const float* rp = adj + (size_t)row * NN;
  float sum = 0.0f;
  for (int k = lane * 4; k < NN; k += 32 * 4) {
    f32x4 v = *(const f32x4*)(rp + k);
    sum += (v[0] + v[1]) + (v[2] + v[3]);
  }
#pragma unroll
  for (int off = 16; off > 0; off >>= 1)
    sum += __shfl_xor(sum, off, 32);
  if (lane == 0) dvec[row] = rsqrtf(sum + 1.0f);
}

// ---------------------------------------------------------------------------
// Kernel 2: s2T[o][n] = d[n] * (x @ W^T)[n][o], stored bf16, feature-major.
// bf16 WMMA GEMM: M=8192 (nodes), N=256 (out feats), K=256.
// Block = 8 waves (2 M x 4 N); wave tile 32x64; block tile 64x256.
// ---------------------------------------------------------------------------
__global__ __launch_bounds__(256) void gcn_supportT_kernel(
    const float* __restrict__ x, const float* __restrict__ W,
    const float* __restrict__ dvec, __bf16* __restrict__ s2T) {
  const int lane = threadIdx.x & 31;
  const int wave = threadIdx.x >> 5;
  const int wm = wave & 1, wn = wave >> 1;
  const int half = lane >> 4, l16 = lane & 15;
  const int mBase = blockIdx.x * 64 + wm * 32;
  const int nBase = wn * 64;

  f32x8 acc[2][4] = {};

  for (int k0 = 0; k0 < DD; k0 += 32) {
    bf16x16 a[2], b[4];
#pragma unroll
    for (int mt = 0; mt < 2; ++mt) {
      // A in this code's contiguous K order (16-bit, 16x32), not the ISA table order: lane half selects K {0..15} vs {16..31}
      const float* p = x + (size_t)(mBase + mt * 16 + l16) * DD + k0 + half * 16;
#pragma unroll
      for (int i = 0; i < 8; ++i) {
        a[mt][i]     = (__bf16)p[i];
        a[mt][i + 8] = (__bf16)p[i + 8];
      }
    }
#pragma unroll
    for (int nt = 0; nt < 4; ++nt) {
      // B in this code's contiguous K order, same as A (16-bit, 32x16): lane = column; 16 contiguous K per lane half
      const float* p = W + (size_t)(nBase + nt * 16 + l16) * DD + k0 + half * 16;
#pragma unroll
      for (int i = 0; i < 16; ++i) b[nt][i] = (__bf16)p[i];
    }
#pragma unroll
    for (int mt = 0; mt < 2; ++mt)
#pragma unroll
      for (int nt = 0; nt < 4; ++nt)
        acc[mt][nt] = __builtin_amdgcn_wmma_f32_16x16x32_bf16(
            false, a[mt], false, b[nt], (short)0, acc[mt][nt], false, false);
  }

  // Epilogue: scale by d[row], convert to bf16, store transposed (contiguous rows)
#pragma unroll
  for (int mt = 0; mt < 2; ++mt) {
    const int r8 = mBase + mt * 16 + half * 8;
    f32x4 dA = *(const f32x4*)(dvec + r8);
    f32x4 dB = *(const f32x4*)(dvec + r8 + 4);
#pragma unroll
    for (int nt = 0; nt < 4; ++nt) {
      const int c = nBase + nt * 16 + l16;
      bf16x8 o;
#pragma unroll
      for (int v = 0; v < 8; ++v) {
        float dv = (v < 4) ? dA[v] : dB[v - 4];
        o[v] = (__bf16)(dv * acc[mt][nt][v]);
      }
      *(bf16x8*)(s2T + (size_t)c * NN + r8) = o;
    }
  }
}

// ---------------------------------------------------------------------------
// Kernel 3: out[i][o] = relu( d[i] * ( sum_j adj[i][j]*s2T[o][j] + s2T[o][i] ) )
// bf16 WMMA GEMM: M=8192, N=256, K=8192. adj converted fp32->bf16 in-register.
// Block tile 64x256 (full N) so adj streams through HBM exactly once.
// ---------------------------------------------------------------------------
__global__ __launch_bounds__(256) void gcn_aggregate_kernel(
    const float* __restrict__ adj, const __bf16* __restrict__ s2T,
    const float* __restrict__ dvec, float* __restrict__ out) {
  const int lane = threadIdx.x & 31;
  const int wave = threadIdx.x >> 5;
  const int wm = wave & 1, wn = wave >> 1;
  const int half = lane >> 4, l16 = lane & 15;
  const int mBase = blockIdx.x * 64 + wm * 32;
  const int nBase = wn * 64;

  f32x8 acc[2][4] = {};

  for (int k0 = 0; k0 < NN; k0 += 32) {
    bf16x16 a[2], b[4];
#pragma unroll
    for (int mt = 0; mt < 2; ++mt) {
      const float* p = adj + (size_t)(mBase + mt * 16 + l16) * NN + k0 + half * 16;
      __builtin_prefetch(p + 32, 0, 1);  // next K tile -> global_prefetch_b8
#pragma unroll
      for (int i = 0; i < 8; ++i) {
        a[mt][i]     = (__bf16)p[i];
        a[mt][i + 8] = (__bf16)p[i + 8];
      }
    }
#pragma unroll
    for (int nt = 0; nt < 4; ++nt) {
      const __bf16* p = s2T + (size_t)(nBase + nt * 16 + l16) * NN + k0 + half * 16;
      bf16x8 lo = *(const bf16x8*)p;
      bf16x8 hi = *(const bf16x8*)(p + 8);
#pragma unroll
      for (int i = 0; i < 8; ++i) {
        b[nt][i]     = lo[i];
        b[nt][i + 8] = hi[i];
      }
    }
#pragma unroll
    for (int mt = 0; mt < 2; ++mt)
#pragma unroll
      for (int nt = 0; nt < 4; ++nt)
        acc[mt][nt] = __builtin_amdgcn_wmma_f32_16x16x32_bf16(
            false, a[mt], false, b[nt], (short)0, acc[mt][nt], false, false);
  }

  // Epilogue: + identity term (s2T[o][i]), scale by d[i], relu, store fp32.
#pragma unroll
  for (int mt = 0; mt < 2; ++mt) {
    const int r8 = mBase + mt * 16 + half * 8;
    f32x4 dA = *(const f32x4*)(dvec + r8);
    f32x4 dB = *(const f32x4*)(dvec + r8 + 4);
#pragma unroll
    for (int nt = 0; nt < 4; ++nt) {
      const int c = nBase + nt * 16 + l16;
      bf16x8 idt = *(const bf16x8*)(s2T + (size_t)c * NN + r8);
#pragma unroll
      for (int v = 0; v < 8; ++v) {
        float dv  = (v < 4) ? dA[v] : dB[v - 4];
        float val = dv * (acc[mt][nt][v] + (float)idt[v]);
        out[(size_t)(r8 + v) * DD + c] = fmaxf(val, 0.0f);
      }
    }
  }
}

// ---------------------------------------------------------------------------
extern "C" void kernel_launch(void* const* d_in, const int* in_sizes, int n_in,
                              void* d_out, int out_size, void* d_ws, size_t ws_size,
                              hipStream_t stream) {
  (void)in_sizes; (void)n_in; (void)out_size; (void)ws_size;
  const float* x   = (const float*)d_in[0];   // [8192, 256]
  const float* adj = (const float*)d_in[1];   // [8192, 8192]
  const float* W   = (const float*)d_in[2];   // [256, 256]
  float* out = (float*)d_out;                 // [8192, 256]

  float*  dvec = (float*)d_ws;                                  // 8192 f32 (32 KB)
  __bf16* s2T  = (__bf16*)((char*)d_ws + NN * sizeof(float));   // 256*8192 bf16 (4 MB)

  gcn_rowsum_kernel  <<<NN / 8, 256, 0, stream>>>(adj, dvec);
  gcn_supportT_kernel<<<NN / 64, 256, 0, stream>>>(x, W, dvec, s2T);
  gcn_aggregate_kernel<<<NN / 64, 256, 0, stream>>>(adj, s2T, dvec, out);
}
